// def_att_5_13864154431693
// MI455X (gfx1250) — matrix operand order repaired and checked by reading, not compiled, not run on hardware
//
#include <hip/hip_runtime.h>
#include <hip/hip_bf16.h>
#include <math.h>

typedef __attribute__((ext_vector_type(16))) __bf16 v16bf;
typedef __attribute__((ext_vector_type(8)))  __bf16 v8bf;
typedef __attribute__((ext_vector_type(8)))  float  v8f;

// ---- problem constants ----
constexpr int BN    = 64;    // batch
constexpr int NPAT  = 256;   // 4*B patches
constexpr int CDIM  = 1024;
constexpr int NGRP  = 8;
constexpr int CGRP  = 128;   // channels per group
constexpr int DHEAD = 64;
constexpr int NHW   = 64;    // 8x8
constexpr int NG2   = 2048;  // NPAT*NGRP
constexpr int CPBH  = 256;

static __device__ __forceinline__ float bf2f(__bf16 v) { return (float)v; }
static __device__ __forceinline__ __bf16 f2bf(float v) { return (__bf16)v; }

static __device__ __forceinline__ v8f wmma_bf16(v16bf a, v16bf b, v8f c) {
  // D = A(16x32) * B(32x16) + C  -> v_wmma_f32_16x16x32_bf16
  return __builtin_amdgcn_wmma_f32_16x16x32_bf16(false, a, false, b, (short)0, c,
                                                 false, false);
}

// A fragment: lane<16 -> elems[0:16]=K[k0..k0+15]
//             lane>=16 -> elems[0:16]=K[k0+16..k0+31]. rowbase points at row*ldk + k0.
static __device__ __forceinline__ v16bf load_frag_a(const __bf16* rowbase, int koff) {
  v8bf lo = *reinterpret_cast<const v8bf*>(rowbase + 2 * koff);
  v8bf hi = *reinterpret_cast<const v8bf*>(rowbase + 2 * koff + 8);
  v16bf r;
#pragma unroll
  for (int e = 0; e < 8; ++e) { r[e] = lo[e]; r[e + 8] = hi[e]; }
  return r;
}

// ---------------- K0: weight conversion ----------------
__global__ void k_convert(const float* wq, const float* w2, const float* outw,
                          __bf16* wq_bf, __bf16* w2t, __bf16* outw_bf) {
  int tid = blockIdx.x * blockDim.x + threadIdx.x;
  int nth = gridDim.x * blockDim.x;
  for (int i = tid; i < NGRP * DHEAD * CGRP; i += nth) wq_bf[i] = f2bf(wq[i]);
  for (int i = tid; i < CPBH * CPBH; i += nth) {
    int o = i / CPBH, in = i % CPBH;          // w2t[out][in] = w2[in][out]
    w2t[i] = f2bf(w2[in * CPBH + o]);
  }
  for (int i = tid; i < CDIM * 512; i += nth) outw_bf[i] = f2bf(outw[i]);
}

// ---------------- K1: LayerNorm + quadrant split -> bf16 [p][hw][c] ----------------
__global__ void k_ln_split(const float* __restrict__ x, const float* __restrict__ g,
                           const float* __restrict__ bb, __bf16* __restrict__ xn_t) {
  __shared__ float ssum[256], ssq[256];
  int blk = blockIdx.x, t = threadIdx.x;
  int b  = blk >> 8;
  int rem = blk & 255;
  int y = rem >> 4, xx = rem & 15;
  float v[4];
  float s = 0.f, q = 0.f;
#pragma unroll
  for (int i = 0; i < 4; ++i) {
    int c = t + i * 256;
    v[i] = x[((size_t)(b * CDIM + c) * 16 + y) * 16 + xx];
    s += v[i]; q += v[i] * v[i];
  }
  ssum[t] = s; ssq[t] = q;
  __syncthreads();
  for (int st = 128; st > 0; st >>= 1) {
    if (t < st) { ssum[t] += ssum[t + st]; ssq[t] += ssq[t + st]; }
    __syncthreads();
  }
  float mu   = ssum[0] * (1.f / 1024.f);
  float var  = ssq[0] * (1.f / 1024.f) - mu * mu;
  float rstd = rsqrtf(var + 1e-6f);
  int qi = y >> 3, qj = xx >> 3, ii = y & 7, jj = xx & 7;
  int p  = (qi * 2 + qj) * BN + b;
  int hw = ii * 8 + jj;
  __bf16* dst = xn_t + (size_t)(p * NHW + hw) * CDIM;
#pragma unroll
  for (int i = 0; i < 4; ++i) {
    int c = t + i * 256;
    dst[c] = f2bf((v[i] - mu) * rstd * g[c] + bb[c]);
  }
}

// ---------------- K2: Q projection via WMMA ----------------
// 8 waves/block, 2 GEMMs/block: wave (wg, mt) owns M-tile mt of GEMM wg,
// all 4 N-tiles; B fragments double-buffered across k-steps.
__global__ void k_qproj(const __bf16* __restrict__ xn_t, const __bf16* __restrict__ wq_bf,
                        float* __restrict__ qbuf) {
  int wave = threadIdx.x >> 5;
  int lane = threadIdx.x & 31;
  int ng = blockIdx.x * 2 + (wave >> 2);
  int mt = wave & 3;
  int n = ng >> 3, g = ng & 7;
  int rowL = lane & 15;
  int koff = (lane >> 4) << 3;   // 0 or 8 (doubled in load_frag_a: A half-lane K offset 0 or 16)
  int bsel = (lane >> 4) << 4;   // 0 or 16 (B half-lane K offset)
  const __bf16* Abase = wq_bf + (size_t)(g * DHEAD + mt * 16 + rowL) * CGRP;
  const __bf16* Bbase = xn_t + (size_t)(n * NHW) * CDIM + g * CGRP;
  v8f acc[4];
#pragma unroll
  for (int nn = 0; nn < 4; ++nn)
#pragma unroll
    for (int k = 0; k < 8; ++k) acc[nn][k] = 0.f;

  v16bf bcur[4], bnxt[4];
#pragma unroll
  for (int nn = 0; nn < 4; ++nn)
    bcur[nn] = *reinterpret_cast<const v16bf*>(
        Bbase + (size_t)(nn * 16 + rowL) * CDIM + bsel);
#pragma unroll
  for (int nn = 0; nn < 4; ++nn) bnxt[nn] = bcur[nn];

  for (int k0 = 0; k0 < CGRP; k0 += 32) {
    v16bf a = load_frag_a(Abase + k0, koff);
    if (k0 + 32 < CGRP) {
#pragma unroll
      for (int nn = 0; nn < 4; ++nn)
        bnxt[nn] = *reinterpret_cast<const v16bf*>(
            Bbase + (size_t)(nn * 16 + rowL) * CDIM + (k0 + 32) + bsel);
    }
#pragma unroll
    for (int nn = 0; nn < 4; ++nn) acc[nn] = wmma_bf16(a, bcur[nn], acc[nn]);
#pragma unroll
    for (int nn = 0; nn < 4; ++nn) bcur[nn] = bnxt[nn];
  }
  float* qout = qbuf + (size_t)ng * DHEAD * NHW;
  int rhalf = (lane >> 4) * 8;
#pragma unroll
  for (int nn = 0; nn < 4; ++nn)
#pragma unroll
    for (int k = 0; k < 8; ++k) {
      int o  = mt * 16 + k + rhalf;
      int hw = nn * 16 + rowL;
      qout[o * NHW + hw] = acc[nn][k];
    }
}

// ---------------- K3: offset conv (depthwise 6x6 s4 p1) + GELU + tanh -> grid ----------------
__global__ void k_offsets(const float* __restrict__ qbuf, const float* __restrict__ w1,
                          const float* __restrict__ b1, const float* __restrict__ w2,
                          float* __restrict__ vg) {
  __shared__ float os[64][4];
  __shared__ float offs[2][4];
  int ng = blockIdx.x, t = threadIdx.x;
  int ch = t >> 2, pos = t & 3;
  int oy = pos >> 1, ox = pos & 1;
  const float* qq = qbuf + (size_t)ng * DHEAD * NHW + ch * NHW;
  float acc = b1[ch];
  for (int ky = 0; ky < 6; ++ky) {
    int iy = oy * 4 - 1 + ky;
    if ((unsigned)iy < 8u)
      for (int kx = 0; kx < 6; ++kx) {
        int ix = ox * 4 - 1 + kx;
        if ((unsigned)ix < 8u) acc += qq[iy * 8 + ix] * w1[ch * 36 + ky * 6 + kx];
      }
  }
  os[ch][pos] = 0.5f * acc * (1.f + erff(acc * 0.70710678118654752f));  // exact GELU
  __syncthreads();
  if (t < 8) {
    int oc = t >> 2, pp = t & 3;
    float d = 0.f;
    for (int c = 0; c < 64; ++c) d += os[c][pp] * w2[oc * 64 + c];
    offs[oc][pp] = tanhf(d) * 4.0f;
  }
  __syncthreads();
  if (t < 4) {
    int pp = t, oyy = pp >> 1, oxx = pp & 1;
    float vx = (float)oxx + offs[0][pp];
    float vy = (float)oyy + offs[1][pp];
    vg[(ng * 4 + pp) * 2 + 0] = 2.f * vx - 1.f;  // hd=wd=2 -> /(2-1)
    vg[(ng * 4 + pp) * 2 + 1] = 2.f * vy - 1.f;
  }
}

// ---------------- K4: bilinear grid sample + K/V proj + sim ----------------
__global__ void k_kv_sim(const __bf16* __restrict__ xn_t, const float* __restrict__ vg,
                         const float* __restrict__ wk, const float* __restrict__ wv,
                         const float* __restrict__ qbuf, float* __restrict__ vbuf,
                         float* __restrict__ simb) {
  __shared__ float kvs[4][128];
  __shared__ float ks[64][4];
  int ng = blockIdx.x, t = threadIdx.x;
  int n = ng >> 3, g = ng & 7;
  const __bf16* img = xn_t + (size_t)(n * NHW) * CDIM + g * CGRP;
  for (int idx = t; idx < 512; idx += 256) {
    int pos = idx >> 7, cin = idx & 127;
    float xs = vg[(ng * 4 + pos) * 2 + 0], ys = vg[(ng * 4 + pos) * 2 + 1];
    float fx = ((xs + 1.f) * 8.f - 1.f) * 0.5f;
    float fy = ((ys + 1.f) * 8.f - 1.f) * 0.5f;
    float x0f = floorf(fx), y0f = floorf(fy);
    int x0 = (int)x0f, y0 = (int)y0f;
    float wx1 = fx - x0f, wy1 = fy - y0f;
    float wx0 = 1.f - wx1, wy0 = 1.f - wy1;
    float acc = 0.f;
#pragma unroll
    for (int dy = 0; dy < 2; ++dy) {
      int yy = y0 + dy;
      if ((unsigned)yy < 8u) {
        float wy = dy ? wy1 : wy0;
#pragma unroll
        for (int dx = 0; dx < 2; ++dx) {
          int xx = x0 + dx;
          if ((unsigned)xx < 8u) {
            float wx = dx ? wx1 : wx0;
            acc += wy * wx * bf2f(img[(size_t)(yy * 8 + xx) * CDIM + cin]);
          }
        }
      }
    }
    kvs[pos][cin] = acc;
  }
  __syncthreads();
  {
    int o = t >> 2, pos = t & 3;
    const float* wkr = wk + (size_t)(g * DHEAD + o) * CGRP;
    const float* wvr = wv + (size_t)(g * DHEAD + o) * CGRP;
    float ka = 0.f, va = 0.f;
    for (int c = 0; c < 128; ++c) { float kvv = kvs[pos][c]; ka += kvv * wkr[c]; va += kvv * wvr[c]; }
    ks[o][pos] = ka;
    vbuf[(size_t)ng * 256 + pos * 64 + o] = va;
  }
  __syncthreads();
  {
    int hw = t >> 2, pos = t & 3;
    const float* qq = qbuf + (size_t)ng * DHEAD * NHW;
    float s = 0.f;
    for (int o = 0; o < 64; ++o) s += qq[o * NHW + hw] * ks[o][pos];
    simb[(size_t)ng * 256 + hw * 4 + pos] = s * 0.125f;  // 64^-0.5
  }
}

// ---------------- K5: CPB MLP ----------------
// 256 threads = 8 waves; 64 rows/block. Wave (nh = w>>2, mt = w&3) owns
// M-tile mt (16 rows) x N-half nh (8 tiles = 128 cols) -> acc[8] (64 VGPRs).
// Per k-step: clause of 8 B-fragment loads, then 8 chained WMMAs.
__global__ void k_cpb(const float* __restrict__ vg, const float* __restrict__ w1,
                      const float* __restrict__ b1, const __bf16* __restrict__ w2t,
                      const float* __restrict__ b2, const float* __restrict__ w3,
                      const float* __restrict__ b3, float* __restrict__ biasb) {
  __shared__ __bf16 h1s[64 * 256];   // 32 KB
  __shared__ float prt[2][64];
  int t = threadIdx.x;
  int r0 = blockIdx.x * 64;
  for (int idx = t; idx < 64 * 256; idx += 256) {
    int lr = idx >> 8, col = idx & 255;
    int r = r0 + lr;
    int ng = r >> 8, hw = (r >> 2) & 63, j = r & 3;
    float gx = 2.f * (float)(hw & 7) * (1.f / 7.f) - 1.f;
    float gy = 2.f * (float)(hw >> 3) * (1.f / 7.f) - 1.f;
    float px = gx - vg[(ng * 4 + j) * 2 + 0];
    float py = gy - vg[(ng * 4 + j) * 2 + 1];
    float f0 = copysignf(log1pf(fabsf(px)), px);
    float f1 = copysignf(log1pf(fabsf(py)), py);
    float h = f0 * w1[col] + f1 * w1[256 + col] + b1[col];
    h1s[lr * 256 + col] = f2bf(fmaxf(h, 0.f));
  }
  __syncthreads();
  int wave = t >> 5, lane = t & 31;
  int nh = wave >> 2, mt = wave & 3;
  int rowL = lane & 15;
  int koff = (lane >> 4) << 3;
  int bsel = (lane >> 4) << 4;
  const __bf16* Wbase = w2t + (size_t)(nh * 128 + rowL) * 256;
  v8f acc[8];
#pragma unroll
  for (int i = 0; i < 8; ++i)
#pragma unroll
    for (int k = 0; k < 8; ++k) acc[i][k] = 0.f;

  for (int k0 = 0; k0 < 256; k0 += 32) {
    if (k0 + 32 < 256)
      __builtin_prefetch(Wbase + (k0 + 32) + bsel, 0, 1);
    v16bf bf[8];
#pragma unroll
    for (int nt = 0; nt < 8; ++nt)
      bf[nt] = *reinterpret_cast<const v16bf*>(Wbase + (size_t)nt * 16 * 256 + k0 + bsel);
    v16bf a = load_frag_a(h1s + (mt * 16 + rowL) * 256 + k0, koff);
#pragma unroll
    for (int nt = 0; nt < 8; ++nt) acc[nt] = wmma_bf16(a, bf[nt], acc[nt]);
  }
  // layer3 partial: sum over this wave's 128 cols of relu(acc+b2)*w3
  float part[8];
#pragma unroll
  for (int k = 0; k < 8; ++k) part[k] = 0.f;
#pragma unroll
  for (int nt = 0; nt < 8; ++nt) {
    int col = nh * 128 + nt * 16 + rowL;
    float b2v = b2[col], w3v = w3[col];
#pragma unroll
    for (int k = 0; k < 8; ++k) part[k] += fmaxf(acc[nt][k] + b2v, 0.f) * w3v;
  }
#pragma unroll
  for (int m = 1; m < 16; m <<= 1)
#pragma unroll
    for (int k = 0; k < 8; ++k) part[k] += __shfl_xor(part[k], m, 32);
  if (rowL == 0) {
    int rbase = mt * 16 + ((lane >> 4) << 3);
#pragma unroll
    for (int k = 0; k < 8; ++k) prt[nh][rbase + k] = part[k];
  }
  __syncthreads();
  if (t < 64) biasb[r0 + t] = prt[0][t] + prt[1][t] + b3[0];
}

// ---------------- K6: softmax(sim + bias) and attn @ V -> bf16 [p][hw][c] ----------------
__global__ void k_softmax_av(const float* __restrict__ simb, const float* __restrict__ biasb,
                             const float* __restrict__ vbuf, __bf16* __restrict__ att_t) {
  __shared__ float attn[64][4];
  __shared__ float vs[256];
  int ng = blockIdx.x, t = threadIdx.x;
  int p = ng >> 3, g = ng & 7;
  vs[t] = vbuf[(size_t)ng * 256 + t];
  attn[t >> 2][t & 3] = simb[(size_t)ng * 256 + t] + biasb[(size_t)ng * 256 + t];
  __syncthreads();
  int hw = t >> 2, j = t & 3;
  float l0 = attn[hw][0], l1 = attn[hw][1], l2 = attn[hw][2], l3 = attn[hw][3];
  float m = fmaxf(fmaxf(l0, l1), fmaxf(l2, l3));
  float e0 = expf(l0 - m), e1 = expf(l1 - m), e2 = expf(l2 - m), e3 = expf(l3 - m);
  float inv = 1.f / (e0 + e1 + e2 + e3);
  float mine = (j == 0 ? e0 : j == 1 ? e1 : j == 2 ? e2 : e3) * inv;
  __syncthreads();
  attn[hw][j] = mine;
  __syncthreads();
  __bf16* dst = att_t + (size_t)p * NHW * 512 + g * 64;
  for (int idx = t; idx < 4096; idx += 256) {
    int hw2 = idx >> 6, o = idx & 63;
    float s = attn[hw2][0] * vs[0 + o] + attn[hw2][1] * vs[64 + o] +
              attn[hw2][2] * vs[128 + o] + attn[hw2][3] * vs[192 + o];
    dst[(size_t)hw2 * 512 + o] = f2bf(s);
  }
}

// ---------------- K7: output projection WMMA + bias + quadrant stitch ----------------
__global__ void k_outproj(const __bf16* __restrict__ att_t, const __bf16* __restrict__ outw_bf,
                          const float* __restrict__ ob, float* __restrict__ out) {
  int blk = blockIdx.x;
  int p = blk >> 3, mc = blk & 7;
  int wave = threadIdx.x >> 5, lane = threadIdx.x & 31;
  int rowL = lane & 15;
  int koff = (lane >> 4) << 3;
  int bsel = (lane >> 4) << 4;
  int rhalf = (lane >> 4) * 8;
  int o0 = mc * 128 + wave * 16;
  const __bf16* Abase = outw_bf + (size_t)(o0 + rowL) * 512;
  const __bf16* Bbase = att_t + (size_t)p * NHW * 512;
  v8f acc[4];
#pragma unroll
  for (int i = 0; i < 4; ++i)
#pragma unroll
    for (int k = 0; k < 8; ++k) acc[i][k] = 0.f;

  v16bf bcur[4], bnxt[4];
#pragma unroll
  for (int nt = 0; nt < 4; ++nt)
    bcur[nt] = *reinterpret_cast<const v16bf*>(
        Bbase + (size_t)(nt * 16 + rowL) * 512 + bsel);
#pragma unroll
  for (int nt = 0; nt < 4; ++nt) bnxt[nt] = bcur[nt];

  for (int k0 = 0; k0 < 512; k0 += 32) {
    v16bf a = load_frag_a(Abase + k0, koff);
    if (k0 + 32 < 512) {
      __builtin_prefetch(Abase + (k0 + 32) + koff, 0, 1);
#pragma unroll
      for (int nt = 0; nt < 4; ++nt)
        bnxt[nt] = *reinterpret_cast<const v16bf*>(
            Bbase + (size_t)(nt * 16 + rowL) * 512 + (k0 + 32) + bsel);
    }
#pragma unroll
    for (int nt = 0; nt < 4; ++nt) acc[nt] = wmma_bf16(a, bcur[nt], acc[nt]);
#pragma unroll
    for (int nt = 0; nt < 4; ++nt) bcur[nt] = bnxt[nt];
  }
  int qi = p >> 7, qj = (p >> 6) & 1, b = p & 63;
#pragma unroll
  for (int nt = 0; nt < 4; ++nt) {
    int hw = nt * 16 + rowL;
    int y = qi * 8 + (hw >> 3);
    int xx = qj * 8 + (hw & 7);
#pragma unroll
    for (int k = 0; k < 8; ++k) {
      int o = o0 + k + rhalf;
      out[((size_t)(b * CDIM + o) * 16 + y) * 16 + xx] = acc[nt][k] + ob[o];
    }
  }
}

// ---------------- host launch ----------------
extern "C" void kernel_launch(void* const* d_in, const int* in_sizes, int n_in,
                              void* d_out, int out_size, void* d_ws, size_t ws_size,
                              hipStream_t stream) {
  (void)in_sizes; (void)n_in; (void)out_size; (void)ws_size;
  const float* x      = (const float*)d_in[0];
  const float* ln_g   = (const float*)d_in[1];
  const float* ln_b   = (const float*)d_in[2];
  const float* wq     = (const float*)d_in[3];
  const float* wk     = (const float*)d_in[4];
  const float* wv     = (const float*)d_in[5];
  const float* off_w1 = (const float*)d_in[6];
  const float* off_b1 = (const float*)d_in[7];
  const float* off_w2 = (const float*)d_in[8];
  const float* cpb_w1 = (const float*)d_in[9];
  const float* cpb_b1 = (const float*)d_in[10];
  const float* cpb_w2 = (const float*)d_in[11];
  const float* cpb_b2 = (const float*)d_in[12];
  const float* cpb_w3 = (const float*)d_in[13];
  const float* cpb_b3 = (const float*)d_in[14];
  const float* out_w  = (const float*)d_in[15];
  const float* out_b  = (const float*)d_in[16];
  float* out = (float*)d_out;

  char* ws = (char*)d_ws;
  size_t off = 0;
  auto carve = [&](size_t bytes) -> char* {
    char* pp = ws + off;
    off += (bytes + 255) & ~(size_t)255;
    return pp;
  };
  __bf16* xn_t    = (__bf16*)carve((size_t)NPAT * NHW * CDIM * 2);
  float*  qbuf    = (float*) carve((size_t)NG2 * DHEAD * NHW * 4);
  __bf16* wq_bf   = (__bf16*)carve((size_t)NGRP * DHEAD * CGRP * 2);
  __bf16* w2t     = (__bf16*)carve((size_t)CPBH * CPBH * 2);
  __bf16* outw_bf = (__bf16*)carve((size_t)CDIM * 512 * 2);
  float*  vg      = (float*) carve((size_t)NG2 * 4 * 2 * 4);
  float*  vbuf    = (float*) carve((size_t)NG2 * 256 * 4);
  float*  simb    = (float*) carve((size_t)NG2 * 256 * 4);
  float*  biasb   = (float*) carve((size_t)NG2 * 256 * 4);
  __bf16* att_t   = (__bf16*)carve((size_t)NPAT * NHW * 512 * 2);

  k_convert<<<64, 256, 0, stream>>>(wq, cpb_w2, out_w, wq_bf, w2t, outw_bf);
  k_ln_split<<<BN * 256, 256, 0, stream>>>(x, ln_g, ln_b, xn_t);
  k_qproj<<<NG2 / 2, 256, 0, stream>>>(xn_t, wq_bf, qbuf);
  k_offsets<<<NG2, 256, 0, stream>>>(qbuf, off_w1, off_b1, off_w2, vg);
  k_kv_sim<<<NG2, 256, 0, stream>>>(xn_t, vg, wk, wv, qbuf, vbuf, simb);
  k_cpb<<<(NG2 * 256) / 64, 256, 0, stream>>>(vg, cpb_w1, cpb_b1, w2t, cpb_b2,
                                              cpb_w3, cpb_b3, biasb);
  k_softmax_av<<<NG2, 256, 0, stream>>>(simb, biasb, vbuf, att_t);
  k_outproj<<<NPAT * 8, 256, 0, stream>>>(att_t, outw_bf, out_b, out);
}
